// Llama4TextMoe_9483287789698
// MI455X (gfx1250) — matrix operand order repaired and checked by reading, not compiled, not run on hardware
//
#include <hip/hip_runtime.h>
#include <hip/hip_bf16.h>
#include <stdint.h>

// ---------------- problem constants ----------------
static constexpr int T      = 2048;   // B*S tokens
static constexpr int H      = 1024;   // hidden
static constexpr int F      = 2048;   // intermediate
static constexpr int TWO_F  = 4096;
static constexpr int E      = 16;     // experts

static constexpr int LDS_STRIDE = 40; // ushorts per row (32 data + 8 pad), 16B aligned

// ---------------- vector / fragment types ----------------
typedef __bf16        v16bf __attribute__((ext_vector_type(16)));
typedef float         v8f   __attribute__((ext_vector_type(8)));
typedef unsigned int  u32x4 __attribute__((ext_vector_type(4)));
typedef float         f32x4 __attribute__((ext_vector_type(4)));

union FragU { v16bf v; u32x4 q[2]; };

__device__ __forceinline__ unsigned short f2bf(float f) {
  unsigned int u = __float_as_uint(f);
  u += 0x7fffu + ((u >> 16) & 1u);          // round-to-nearest-even
  return (unsigned short)(u >> 16);
}

__device__ __forceinline__ float silu(float x) {
  return x / (1.0f + __expf(-x));
}

__device__ __forceinline__ v8f wmma_bf16(v16bf a, v16bf b, v8f c) {
  return __builtin_amdgcn_wmma_f32_16x16x32_bf16(
      /*neg_a=*/false, a, /*neg_b=*/false, b,
      /*c_mod=*/(short)0, c, /*reuse_a=*/false, /*reuse_b=*/false);
}

// A fragment, 16x32 bf16, from LDS tile [16][LDS_STRIDE] (row-major over K).
// ISA layout: lane<16 -> M=lane, halves = K0..7,K16..23 ; lane>=16 -> M=lane-16, K8..15,K24..31
__device__ __forceinline__ v16bf load_frag_a(const unsigned short* lds, int lane) {
  int m  = lane & 15;
  int kb = (lane & 16) ? 8 : 0;
  FragU f;
  f.q[0] = *(const u32x4*)(lds + m * LDS_STRIDE + kb);
  f.q[1] = *(const u32x4*)(lds + m * LDS_STRIDE + kb + 16);
  return f.v;
}

// B fragment, 32x16 bf16, from LDS tile stored transposed: [n][k] rows of LDS_STRIDE.
// ISA layout: lane<16 -> N=lane, K0..7,K16..23 ; lane>=16 -> N=lane-16, K8..15,K24..31
__device__ __forceinline__ v16bf load_frag_b(const unsigned short* lds, int lane) {
  int n  = lane & 15;
  int kb = (lane & 16) ? 8 : 0;
  FragU f;
  f.q[0] = *(const u32x4*)(lds + n * LDS_STRIDE + kb);
  f.q[1] = *(const u32x4*)(lds + n * LDS_STRIDE + kb + 16);
  return f.v;
}

// ---------------- pipelined tile loaders (256 threads): fetch->regs, commit->LDS ----

struct ARegs { u32x4 d0, d1; };

// A tile: 128 rows x 32 bf16 from bf16 source (row = toks[row], -1 -> zero fill)
__device__ __forceinline__ ARegs a_fetch(const unsigned short* __restrict__ src,
                                         int srcStride, int k0,
                                         const int* __restrict__ toks) {
  int row = threadIdx.x >> 1;
  int seg = (threadIdx.x & 1) * 16;
  ARegs r; r.d0 = (u32x4)0u; r.d1 = (u32x4)0u;
  int t = toks[row];
  if (t >= 0) {
    const u32x4* p = (const u32x4*)(src + (size_t)t * srcStride + k0 + seg);
    r.d0 = p[0]; r.d1 = p[1];
  }
  return r;
}
__device__ __forceinline__ void a_commit(ARegs r, unsigned short* __restrict__ lds) {
  int row = threadIdx.x >> 1;
  int seg = (threadIdx.x & 1) * 16;
  u32x4* q = (u32x4*)(lds + row * LDS_STRIDE + seg);
  q[0] = r.d0; q[1] = r.d1;
}

struct BRegs { f32x4 a, b; };

// B direct: weight memory is [n][k] (rowLen = k length). 64 n rows x 32 k.
__device__ __forceinline__ BRegs bdir_fetch(const float* __restrict__ w, int rowLen,
                                            int n0, int k0) {
  int n   = threadIdx.x >> 2;          // 0..63
  int seg = (threadIdx.x & 3) * 8;     // 0..24
  const f32x4* p = (const f32x4*)(w + (size_t)(n0 + n) * rowLen + k0 + seg);
  BRegs r; r.a = p[0]; r.b = p[1];
  return r;
}
__device__ __forceinline__ void bdir_commit(BRegs r, unsigned short* __restrict__ lds) {
  int n   = threadIdx.x >> 2;
  int seg = (threadIdx.x & 3) * 8;
  union { unsigned short s[8]; u32x4 q; } o;
  o.s[0] = f2bf(r.a.x); o.s[1] = f2bf(r.a.y); o.s[2] = f2bf(r.a.z); o.s[3] = f2bf(r.a.w);
  o.s[4] = f2bf(r.b.x); o.s[5] = f2bf(r.b.y); o.s[6] = f2bf(r.b.z); o.s[7] = f2bf(r.b.w);
  *(u32x4*)(lds + n * LDS_STRIDE + seg) = o.q;
}

// B transposed: weight memory is [k][n] (rowLen = n length). LDS stored as [n][k].
__device__ __forceinline__ BRegs btr_fetch(const float* __restrict__ w, int rowLen,
                                           int k0, int n0) {
  int kk = threadIdx.x >> 3;           // 0..31
  int nn = (threadIdx.x & 7) * 8;      // 0..56
  const f32x4* p = (const f32x4*)(w + (size_t)(k0 + kk) * rowLen + n0 + nn);
  BRegs r; r.a = p[0]; r.b = p[1];
  return r;
}
__device__ __forceinline__ void btr_commit(BRegs r, unsigned short* __restrict__ lds) {
  int kk = threadIdx.x >> 3;
  int nn = (threadIdx.x & 7) * 8;
  unsigned short v[8];
  v[0] = f2bf(r.a.x); v[1] = f2bf(r.a.y); v[2] = f2bf(r.a.z); v[3] = f2bf(r.a.w);
  v[4] = f2bf(r.b.x); v[5] = f2bf(r.b.y); v[6] = f2bf(r.b.z); v[7] = f2bf(r.b.w);
#pragma unroll
  for (int i = 0; i < 8; ++i) lds[(nn + i) * LDS_STRIDE + kk] = v[i];
}

// ---------------- kernel 1: router (top-1 + sigmoid score), also zeroes counts ----------------
__global__ __launch_bounds__(256) void k_router(const float* __restrict__ x,
                                                const float* __restrict__ wr,
                                                int* __restrict__ topk,
                                                float* __restrict__ score,
                                                int* __restrict__ counts) {
  __shared__ float sc[16][17];
  int lt = threadIdx.x >> 4;
  int e  = threadIdx.x & 15;
  int t  = blockIdx.x * 16 + lt;
  if (blockIdx.x == 0 && threadIdx.x < E) counts[threadIdx.x] = 0;

  const f32x4* xa = (const f32x4*)(x + (size_t)t * H);
  const f32x4* wa = (const f32x4*)(wr + (size_t)e * H);
  float acc = 0.f;
#pragma unroll 4
  for (int i = 0; i < H / 4; ++i) {
    f32x4 a = xa[i], w4 = wa[i];
    acc += a.x * w4.x + a.y * w4.y + a.z * w4.z + a.w * w4.w;
  }
  sc[lt][e] = acc;
  __syncthreads();
  if (e == 0) {
    float best = sc[lt][0]; int bi = 0;
#pragma unroll
    for (int j = 1; j < 16; ++j) { float v = sc[lt][j]; if (v > best) { best = v; bi = j; } }
    topk[t]  = bi;
    score[t] = 1.0f / (1.0f + __expf(-best));
  }
}

// ---------------- kernel 2: bucket compaction ----------------
__global__ __launch_bounds__(256) void k_bucket(const int* __restrict__ topk,
                                                int* __restrict__ counts,
                                                int* __restrict__ list) {
  int t = blockIdx.x * 256 + threadIdx.x;
  int e = topk[t];
  int pos = atomicAdd(&counts[e], 1);
  list[e * T + pos] = t;
}

// ---------------- kernel 3: bf16 conversion: xb = bf16(x), xs = bf16(score*x) ----------------
__global__ __launch_bounds__(256) void k_scale(const float* __restrict__ x,
                                               const float* __restrict__ score,
                                               unsigned short* __restrict__ xb,
                                               unsigned short* __restrict__ xs) {
  int gid = blockIdx.x * 256 + threadIdx.x;
  float s = score[gid >> 10];       // H == 1024
  float v = x[gid];
  xb[gid] = f2bf(v);
  xs[gid] = f2bf(v * s);
}

// ---------------- GEMM 1: SwiGLU up-projection (gate & up fused), 128x64 tile ----------------
// GATHER=false: shared MLP  (A=xb, Wg=w_gate [F,H] direct, Wu=w_up [F,H] direct)
// GATHER=true : expert MLP  (A=xs gathered, Wg=gate_up_proj[e] [H,2F] transposed; gate col n, up col F+n)
template <bool GATHER>
__global__ __launch_bounds__(256) void k_swiglu_gemm(const unsigned short* __restrict__ A,
                                                     const float* __restrict__ Wg,
                                                     const float* __restrict__ Wu,
                                                     const int* __restrict__ counts,
                                                     const int* __restrict__ list,
                                                     unsigned short* __restrict__ Hout) {
  int e   = GATHER ? blockIdx.z : 0;
  int cnt = GATHER ? counts[e] : T;
  int m0  = blockIdx.y * 128;
  if (m0 >= cnt) return;
  int n0  = blockIdx.x * 64;      // over F

  __shared__ unsigned short aL[128 * LDS_STRIDE];
  __shared__ unsigned short bgL[64 * LDS_STRIDE];
  __shared__ unsigned short buL[64 * LDS_STRIDE];
  __shared__ int toks[128];

  if (threadIdx.x < 128) {
    int g = m0 + (int)threadIdx.x;
    toks[threadIdx.x] = (g < cnt) ? (GATHER ? list[e * T + g] : g) : -1;
  }
  __syncthreads();

  int lane = threadIdx.x & 31;
  int wave = threadIdx.x >> 5;

  v8f zero = {0.f, 0.f, 0.f, 0.f, 0.f, 0.f, 0.f, 0.f};
  v8f accg[4], accu[4];
#pragma unroll
  for (int j = 0; j < 4; ++j) { accg[j] = zero; accu[j] = zero; }

  const float* wg = GATHER ? (Wg + (size_t)e * H * TWO_F) : Wg;

  // -------- software pipeline: prologue --------
  ARegs ar = a_fetch(A, H, 0, toks);
  BRegs bgr, bur;
  if (GATHER) { bgr = btr_fetch(wg, TWO_F, 0, n0); bur = btr_fetch(wg, TWO_F, 0, F + n0); }
  else        { bgr = bdir_fetch(Wg, H, n0, 0);    bur = bdir_fetch(Wu, H, n0, 0); }
  a_commit(ar, aL);
  if (GATHER) { btr_commit(bgr, bgL); btr_commit(bur, buL); }
  else        { bdir_commit(bgr, bgL); bdir_commit(bur, buL); }
  __syncthreads();

  for (int k0 = 0; k0 < H; k0 += 32) {
    int kn = k0 + 32;
    bool more = kn < H;
    // issue next tile's global loads (latency hidden behind the WMMAs below)
    if (more) {
      ar = a_fetch(A, H, kn, toks);
      if (GATHER) { bgr = btr_fetch(wg, TWO_F, kn, n0); bur = btr_fetch(wg, TWO_F, kn, F + n0); }
      else        { bgr = bdir_fetch(Wg, H, n0, kn);    bur = bdir_fetch(Wu, H, n0, kn); }
    }

    // compute: batch fragment ds_loads, then back-to-back WMMAs
    v16bf a = load_frag_a(aL + wave * 16 * LDS_STRIDE, lane);
    v16bf bg[4], bu[4];
#pragma unroll
    for (int j = 0; j < 4; ++j) bg[j] = load_frag_b(bgL + j * 16 * LDS_STRIDE, lane);
#pragma unroll
    for (int j = 0; j < 4; ++j) accg[j] = wmma_bf16(a, bg[j], accg[j]);
#pragma unroll
    for (int j = 0; j < 4; ++j) bu[j] = load_frag_b(buL + j * 16 * LDS_STRIDE, lane);
#pragma unroll
    for (int j = 0; j < 4; ++j) accu[j] = wmma_bf16(a, bu[j], accu[j]);

    __syncthreads();
    if (more) {
      a_commit(ar, aL);
      if (GATHER) { btr_commit(bgr, bgL); btr_commit(bur, buL); }
      else        { bdir_commit(bgr, bgL); bdir_commit(bur, buL); }
      __syncthreads();
    }
  }

  // epilogue: h = silu(gate) * up -> bf16, scattered by token row
  int ncol0 = n0 + (lane & 15);
  int moff  = (lane & 16) ? 8 : 0;
#pragma unroll
  for (int j = 0; j < 4; ++j) {
#pragma unroll
    for (int r = 0; r < 8; ++r) {
      int tok = toks[wave * 16 + moff + r];
      if (tok >= 0) {
        float hv = silu(accg[j][r]) * accu[j][r];
        Hout[(size_t)tok * F + ncol0 + j * 16] = f2bf(hv);
      }
    }
  }
}

// ---------------- GEMM 2: down projection, 128x64 tile, K = F ----------------
// GATHER=false: out[t][n]   = hs @ w_down.T        (w_down [H,F] direct)
// GATHER=true : out[tok][n] += hx @ down_proj[e]   ([F,H] transposed)
template <bool GATHER>
__global__ __launch_bounds__(256) void k_down_gemm(const unsigned short* __restrict__ A,
                                                   const float* __restrict__ W,
                                                   const int* __restrict__ counts,
                                                   const int* __restrict__ list,
                                                   float* __restrict__ out) {
  int e   = GATHER ? blockIdx.z : 0;
  int cnt = GATHER ? counts[e] : T;
  int m0  = blockIdx.y * 128;
  if (m0 >= cnt) return;
  int n0  = blockIdx.x * 64;      // over H

  __shared__ unsigned short aL[128 * LDS_STRIDE];
  __shared__ unsigned short bL[64 * LDS_STRIDE];
  __shared__ int toks[128];

  if (threadIdx.x < 128) {
    int g = m0 + (int)threadIdx.x;
    toks[threadIdx.x] = (g < cnt) ? (GATHER ? list[e * T + g] : g) : -1;
  }
  __syncthreads();

  int lane = threadIdx.x & 31;
  int wave = threadIdx.x >> 5;

  v8f zero = {0.f, 0.f, 0.f, 0.f, 0.f, 0.f, 0.f, 0.f};
  v8f acc[4];
#pragma unroll
  for (int j = 0; j < 4; ++j) acc[j] = zero;

  const float* w = GATHER ? (W + (size_t)e * F * H) : W;

  // -------- software pipeline: prologue --------
  ARegs ar = a_fetch(A, F, 0, toks);
  BRegs br = GATHER ? btr_fetch(w, H, 0, n0) : bdir_fetch(W, F, n0, 0);
  a_commit(ar, aL);
  if (GATHER) btr_commit(br, bL); else bdir_commit(br, bL);
  __syncthreads();

  for (int k0 = 0; k0 < F; k0 += 32) {
    int kn = k0 + 32;
    bool more = kn < F;
    if (more) {
      ar = a_fetch(A, F, kn, toks);
      br = GATHER ? btr_fetch(w, H, kn, n0) : bdir_fetch(W, F, n0, kn);
    }

    v16bf a = load_frag_a(aL + wave * 16 * LDS_STRIDE, lane);
    v16bf b[4];
#pragma unroll
    for (int j = 0; j < 4; ++j) b[j] = load_frag_b(bL + j * 16 * LDS_STRIDE, lane);
#pragma unroll
    for (int j = 0; j < 4; ++j) acc[j] = wmma_bf16(a, b[j], acc[j]);

    __syncthreads();
    if (more) {
      a_commit(ar, aL);
      if (GATHER) btr_commit(br, bL); else bdir_commit(br, bL);
      __syncthreads();
    }
  }

  int ncol0 = n0 + (lane & 15);
  int moff  = (lane & 16) ? 8 : 0;
#pragma unroll
  for (int j = 0; j < 4; ++j) {
#pragma unroll
    for (int r = 0; r < 8; ++r) {
      int tok = toks[wave * 16 + moff + r];
      if (tok >= 0) {
        size_t idx = (size_t)tok * H + ncol0 + j * 16;
        if (GATHER) out[idx] += acc[j][r];   // expert add (unique writer per element)
        else        out[idx]  = acc[j][r];   // shared MLP writes (idempotent base)
      }
    }
  }
}

// ---------------- host launch ----------------
extern "C" void kernel_launch(void* const* d_in, const int* in_sizes, int n_in,
                              void* d_out, int out_size, void* d_ws, size_t ws_size,
                              hipStream_t stream) {
  const float* x        = (const float*)d_in[0];   // [T,H]
  const float* w_router = (const float*)d_in[1];   // [E,H]
  const float* gate_up  = (const float*)d_in[2];   // [E,H,2F]
  const float* down     = (const float*)d_in[3];   // [E,F,H]
  const float* w_gate   = (const float*)d_in[4];   // [F,H]
  const float* w_up     = (const float*)d_in[5];   // [F,H]
  const float* w_down   = (const float*)d_in[6];   // [H,F]
  float* out = (float*)d_out;

  char* ws = (char*)d_ws;
  size_t off = 0;
  auto alloc = [&](size_t bytes) { void* p = ws + off; off = (off + bytes + 255) & ~(size_t)255; return p; };
  unsigned short* xb   = (unsigned short*)alloc((size_t)T * H * 2);
  unsigned short* xs   = (unsigned short*)alloc((size_t)T * H * 2);
  unsigned short* h_sh = (unsigned short*)alloc((size_t)T * F * 2);
  unsigned short* h_ex = (unsigned short*)alloc((size_t)T * F * 2);
  int*   topk   = (int*)alloc((size_t)T * 4);
  float* score  = (float*)alloc((size_t)T * 4);
  int*   counts = (int*)alloc((size_t)E * 4);
  int*   list   = (int*)alloc((size_t)E * T * 4);
  (void)ws_size; (void)in_sizes; (void)n_in; (void)out_size;

  // 1) router (+ zero counters)
  k_router<<<T / 16, 256, 0, stream>>>(x, w_router, topk, score, counts);
  // 2) per-expert token compaction
  k_bucket<<<T / 256, 256, 0, stream>>>(topk, counts, list);
  // 3) bf16 conversions (x and score*x)
  k_scale<<<(T * H) / 256, 256, 0, stream>>>(x, score, xb, xs);
  // 4) shared SwiGLU: hs = silu(x@w_gate.T) * (x@w_up.T)
  k_swiglu_gemm<false><<<dim3(F / 64, T / 128, 1), 256, 0, stream>>>(
      xb, w_gate, w_up, nullptr, nullptr, h_sh);
  // 5) shared down: out = hs @ w_down.T   (full overwrite -> idempotent base)
  k_down_gemm<false><<<dim3(H / 64, T / 128, 1), 256, 0, stream>>>(
      h_sh, w_down, nullptr, nullptr, out);
  // 6) expert SwiGLU on gathered, score-scaled tokens
  k_swiglu_gemm<true><<<dim3(F / 64, T / 128, E), 256, 0, stream>>>(
      xs, gate_up, nullptr, counts, list, h_ex);
  // 7) expert down: out += h_ex @ down_proj[e]
  k_down_gemm<true><<<dim3(H / 64, T / 128, E), 256, 0, stream>>>(
      h_ex, down, counts, list, out);
}
